// MahalanobisDistanceMetric_64029372449012
// MI455X (gfx1250) — matrix operand order repaired and checked by reading, not compiled, not run on hardware
//
#include <hip/hip_runtime.h>
#include <hip/hip_bf16.h>

// Mahalanobis distance metric for MI455X (gfx1250).
// B=64, C=256, H=W=56 -> P=3136.
// Per pixel: T = Delta(64x256) @ M_p(256x256) via v_wmma_f32_16x16x32_bf16,
// md[b] = sum_c T[b,c]*Delta[b,c], out = mean(sqrt(md)).
//
// Roofline: inv_cov (822 MB, streamed once, NT hint) dominates -> ~44 us at
// 23.3 TB/s. 26.3 GFLOP in bf16-WMMA is far below the matrix ceiling, so the
// kernel is memory-bound; inv_cov goes straight from global to B-fragments.

typedef __bf16 bf16_t;
typedef __attribute__((ext_vector_type(16))) __bf16 v16bf;
typedef __attribute__((ext_vector_type(8)))  __bf16 v8bf;
typedef __attribute__((ext_vector_type(8)))  float  v8f;

#define B_DIM 64
#define C_DIM 256
#define P_DIM 3136
#define LDSS  264   // bf16 row stride: 264*2 = 528 bytes, multiple of 16

// Sum over each 16-lane half via ds_swizzle xor butterflies
// (group-of-32 mode: offset = xor_mask<<10 | and_mask 0x1F, or_mask 0).
// After this, every lane in a half holds that half's sum.
__device__ __forceinline__ float half16_reduce_add(float v) {
    v += __int_as_float(__builtin_amdgcn_ds_swizzle(__float_as_int(v), (1 << 10) | 0x1F));
    v += __int_as_float(__builtin_amdgcn_ds_swizzle(__float_as_int(v), (2 << 10) | 0x1F));
    v += __int_as_float(__builtin_amdgcn_ds_swizzle(__float_as_int(v), (4 << 10) | 0x1F));
    v += __int_as_float(__builtin_amdgcn_ds_swizzle(__float_as_int(v), (8 << 10) | 0x1F));
    return v;
}

__global__ __launch_bounds__(256) void mahal_pixel_kernel(
    const float* __restrict__ emb,     // [B, C, P]
    const float* __restrict__ mean,    // [C, P]
    const float* __restrict__ invcov,  // [P, C, C]
    float* __restrict__ partial)       // [P] per-pixel sum of 64 sqrt distances
{
    __shared__ __align__(16) __bf16 sDelta[B_DIM][LDSS];   // 33792 B
    __shared__ float sWavePart[8][B_DIM + 1];              // per-wave md partials
    __shared__ float sDist[B_DIM];

    const int p    = blockIdx.x;
    const int tid  = threadIdx.x;
    const int lane = tid & 31;
    const int wave = tid >> 5;
    const int hi   = lane >> 4;   // 0: lanes 0-15, 1: lanes 16-31
    const int l16  = lane & 15;

    // ---------- Phase 1: delta = emb - mean, staged to LDS as bf16 ----------
    // thread tid owns channel c = tid; loops over all 64 batches. The stride-P
    // gather is a transpose; adjacent-pixel workgroups share cachelines and the
    // 192 MB L2 (kept clean by NT inv_cov loads) absorbs the replication.
    {
        const int c = tid;
        const float m = mean[(size_t)c * P_DIM + p];
        const float* ep = emb + (size_t)c * P_DIM + p;
        #pragma unroll 8
        for (int b = 0; b < B_DIM; ++b) {
            float d = ep[(size_t)b * C_DIM * P_DIM] - m;
            sDelta[b][c] = (__bf16)d;
        }
    }
    __syncthreads();

    // ---------- Phase 2: T = Delta @ M_p via WMMA ----------
    // Wave w owns N-tiles {2w, 2w+1}; all 4 M-tiles (b rows).
    const int nt0 = wave * 2;
    const size_t Mbase = (size_t)p * C_DIM * C_DIM;

    v8f acc[4][2];
    #pragma unroll
    for (int mt = 0; mt < 4; ++mt)
        #pragma unroll
        for (int nt = 0; nt < 2; ++nt)
            acc[mt][nt] = (v8f){0.f, 0.f, 0.f, 0.f, 0.f, 0.f, 0.f, 0.f};

    for (int kk = 0; kk < 8; ++kk) {           // K = 256 in steps of 32
        // A fragments (contiguous k order, same as B): lane holds K
        // [ka .. ka+15], ka = kk*32 + 16*hi, row = 16*mt + l16.
        v16bf afrag[4];
        const int ka = kk * 32 + hi * 16;
        #pragma unroll
        for (int mt = 0; mt < 4; ++mt) {
            const __bf16* rowp = &sDelta[mt * 16 + l16][0];
            v8bf lo = *(const v8bf*)(rowp + ka);
            v8bf hh = *(const v8bf*)(rowp + ka + 8);
            afrag[mt] = __builtin_shufflevector(lo, hh,
                0, 1, 2, 3, 4, 5, 6, 7, 8, 9, 10, 11, 12, 13, 14, 15);
        }

        #pragma unroll
        for (int nt = 0; nt < 2; ++nt) {
            // B fragment (contiguous k order, same as A): lane column n, 16 consecutive K rows
            // starting at kk*32 + 16*hi. inv_cov is streamed once -> NT loads
            // keep L2 capacity for the embeddings transpose reuse.
            const int n    = (nt0 + nt) * 16 + l16;
            const int krow = kk * 32 + hi * 16;
            const float* bp = invcov + Mbase + (size_t)krow * C_DIM + n;
            v16bf bfrag;
            #pragma unroll
            for (int i = 0; i < 16; ++i)
                bfrag[i] = (__bf16)__builtin_nontemporal_load(bp + (size_t)i * C_DIM);

            #pragma unroll
            for (int mt = 0; mt < 4; ++mt) {
                acc[mt][nt] = __builtin_amdgcn_wmma_f32_16x16x32_bf16(
                    /*neg_a=*/false, afrag[mt],
                    /*neg_b=*/false, bfrag,
                    /*c_mod=*/(short)0, acc[mt][nt],
                    /*reuse_a=*/false, /*reuse_b=*/false);
            }
        }
    }

    // ---------- Phase 3: md[b] = sum_c T[b,c] * delta[b,c] ----------
    // C/D layout: elem v -> b = 16*mt + v + 8*hi, column = nt*16 + l16.
    #pragma unroll
    for (int mt = 0; mt < 4; ++mt) {
        #pragma unroll
        for (int v = 0; v < 8; ++v) {
            const int brow = mt * 16 + v + hi * 8;
            float val = 0.f;
            #pragma unroll
            for (int nt = 0; nt < 2; ++nt) {
                const int n = (nt0 + nt) * 16 + l16;
                val += acc[mt][nt][v] * (float)sDelta[brow][n];
            }
            val = half16_reduce_add(val);
            if (l16 == 0) sWavePart[wave][brow] = val;
        }
    }
    __syncthreads();

    // Deterministic fixed-order reduction across waves, then sqrt.
    if (tid < B_DIM) {
        float md = 0.f;
        #pragma unroll
        for (int w = 0; w < 8; ++w) md += sWavePart[w][tid];
        sDist[tid] = sqrtf(fmaxf(md, 0.f));
    }
    __syncthreads();

    if (tid == 0) {
        float s = 0.f;
        #pragma unroll 8
        for (int b = 0; b < B_DIM; ++b) s += sDist[b];
        partial[p] = s;
    }
}

__global__ __launch_bounds__(256) void mahal_reduce_kernel(
    const float* __restrict__ partial, float* __restrict__ out,
    int n, float scale)
{
    __shared__ float s[256];
    float v = 0.f;
    for (int i = threadIdx.x; i < n; i += 256) v += partial[i];  // fixed order
    s[threadIdx.x] = v;
    __syncthreads();
    if (threadIdx.x == 0) {
        float t = 0.f;
        for (int i = 0; i < 256; ++i) t += s[i];                 // fixed order
        out[0] = t * scale;
    }
}

extern "C" void kernel_launch(void* const* d_in, const int* in_sizes, int n_in,
                              void* d_out, int out_size, void* d_ws, size_t ws_size,
                              hipStream_t stream) {
    const float* emb    = (const float*)d_in[0];  // [64, 256, 56, 56]
    const float* mean   = (const float*)d_in[1];  // [256, 3136]
    const float* invcov = (const float*)d_in[2];  // [3136, 256, 256]
    float* out = (float*)d_out;
    float* ws  = (float*)d_ws;                    // >= 3136 floats

    mahal_pixel_kernel<<<P_DIM, 256, 0, stream>>>(emb, mean, invcov, ws);
    mahal_reduce_kernel<<<1, 256, 0, stream>>>(
        ws, out, P_DIM, 1.0f / ((float)B_DIM * (float)P_DIM));
}
